// T5Attention_33483565040080
// MI455X (gfx1250) — matrix operand order repaired and checked by reading, not compiled, not run on hardware
//
#include <hip/hip_runtime.h>
#include <math.h>

// ---------------------------------------------------------------------------
// T5 self-attention for MI455X (gfx1250). BF16 WMMA everywhere, f32 accum.
// B=4, S=2048, H=16, D=64, d_model=1024.
// ---------------------------------------------------------------------------

#define BATCH   4
#define SEQ     2048
#define HEADS   16
#define DKV     64
#define DMODEL  1024
#define NBUCK   32
#define RELSPAN 4095   // rel in [-2047, 2047]

// GEMM dims (all four projections share these)
#define GM 8192
#define GN 1024
#define GK 1024

typedef __attribute__((ext_vector_type(16))) __bf16 bf16x16;
typedef __attribute__((ext_vector_type(8)))  __bf16 bf16x8;
typedef __attribute__((ext_vector_type(8)))  float  f32x8;

__device__ __forceinline__ __bf16 f2bf16(float f) {
  unsigned u = __builtin_bit_cast(unsigned, f);
  u += 0x7FFFu + ((u >> 16) & 1u);          // round-to-nearest-even
  return __builtin_bit_cast(__bf16, (unsigned short)(u >> 16));
}

__device__ __forceinline__ f32x8 wmma_bf16(bf16x16 a, bf16x16 b, f32x8 c) {
  return __builtin_amdgcn_wmma_f32_16x16x32_bf16(
      /*neg_a=*/false, a, /*neg_b=*/false, b,
      /*c_mod=*/(short)0, c, /*reuse_a=*/false, /*reuse_b=*/false);
}

// A-fragment (16x32 bf16): lane<16 -> row, K0..15;
// lane>=16 -> same row, K16..31. Two 16B chunks per lane.
__device__ __forceinline__ bf16x16 load_a_frag(const __bf16* row, int halfid) {
  bf16x8 c0 = *reinterpret_cast<const bf16x8*>(row + halfid * 16);
  bf16x8 c1 = *reinterpret_cast<const bf16x8*>(row + halfid * 16 + 8);
  return __builtin_shufflevector(c0, c1, 0,1,2,3,4,5,6,7,8,9,10,11,12,13,14,15);
}

// CDNA5 async copy global->LDS (16B per lane), tracked by ASYNCcnt.
// LDS byte offset = low 32 bits of the generic pointer (LDS aperture mapping).
__device__ __forceinline__ void async_load_b128(void* lds_dst, const void* gsrc) {
  unsigned lds_off = (unsigned)(unsigned long long)lds_dst;
  asm volatile("global_load_async_to_lds_b128 %0, %1, off"
               :: "v"(lds_off), "v"(gsrc)
               : "memory");
}
__device__ __forceinline__ void async_wait0() {
  asm volatile("s_wait_asynccnt 0" ::: "memory");
}

// ---------------------------------------------------------------------------
// f32 -> bf16 conversion (grid-stride)
// ---------------------------------------------------------------------------
__global__ void k_f32_to_bf16(const float* __restrict__ in,
                              __bf16* __restrict__ out, int n) {
  for (int i = blockIdx.x * blockDim.x + threadIdx.x; i < n;
       i += gridDim.x * blockDim.x)
    out[i] = f2bf16(in[i]);
}

// ---------------------------------------------------------------------------
// T5 relative-position bucket table:  relTable[h][rel+2047]
// ---------------------------------------------------------------------------
__device__ __forceinline__ int rel_bucket(int rel) {
  const int nb = NBUCK / 2;         // 16
  const int max_exact = nb / 2;     // 8
  int ret = (rel > 0) ? nb : 0;
  int rp  = rel < 0 ? -rel : rel;
  if (rp < max_exact) return ret + rp;
  float lf = __logf((float)rp * (1.0f / 8.0f)) * (8.0f / __logf(16.0f));
  int large = max_exact + (int)lf;
  if (large > nb - 1) large = nb - 1;
  return ret + large;
}

__global__ void k_build_rel_table(const float* __restrict__ tbl,  // [32,16]
                                  float* __restrict__ relTable) { // [16,4095]
  int idx = blockIdx.x * blockDim.x + threadIdx.x;
  if (idx >= HEADS * RELSPAN) return;
  int h   = idx / RELSPAN;
  int rel = idx % RELSPAN - 2047;
  relTable[idx] = tbl[rel_bucket(rel) * HEADS + h];
}

// Write pos_bias output [1,16,2048,2048]; 4 k-values per thread.
__global__ void k_write_pos_bias(const float* __restrict__ relTable,
                                 float* __restrict__ outBias) {
  long long i = (long long)blockIdx.x * blockDim.x + threadIdx.x;
  long long idx = i * 4;                         // element index, < 16*2^22
  int h   = (int)(idx >> 22);
  int rem = (int)(idx & ((1 << 22) - 1));
  int q   = rem >> 11;
  int k   = rem & 2047;
  const float* row = relTable + h * RELSPAN + 2047 - q;
  float4 v;
  v.x = row[k + 0]; v.y = row[k + 1]; v.z = row[k + 2]; v.w = row[k + 3];
  *reinterpret_cast<float4*>(outBias + idx) = v;
}

// ---------------------------------------------------------------------------
// Tiled BF16 WMMA GEMM: C[GM,GN] = A[GM,GK] * B[GK,GN].  OutT = float | __bf16.
// 256 threads = 8 waves; WG tile 128x128; wave tile 32x64; K step 32.
// Double-buffered LDS: async global->LDS copy of tile t+1 overlaps the 8 WMMAs
// of tile t; one s_wait_asynccnt + one barrier per K step.
// ---------------------------------------------------------------------------
template <typename OutT>
__global__ void __launch_bounds__(256)
k_gemm_tpl(const __bf16* __restrict__ A, const __bf16* __restrict__ Bw,
           OutT* __restrict__ C) {
  __shared__ __align__(32) __bf16 Alds[2][128][32];
  __shared__ __align__(32) __bf16 Bt[2][128][32];   // transposed: Bt[col][k]

  const int m0 = blockIdx.x * 128, n0 = blockIdx.y * 128;
  const int t = threadIdx.x, wave = t >> 5, lane = t & 31;
  const int mw = (wave >> 1) * 32, nw = (wave & 1) * 64;
  const int halfid = lane >> 4, l16 = lane & 15;

  // staging assignments (fixed per thread)
  const int ar = t >> 1, ach = (t & 1) * 16;     // A: row 0..127, col chunk
  const int bkr = t >> 3, bcc = (t & 7) * 16;    // B: k-row 0..31, col chunk

  auto stage = [&](int buf, int k0) {
    const __bf16* asrc = A + (size_t)(m0 + ar) * GK + k0 + ach;
    async_load_b128(&Alds[buf][ar][ach], asrc);
    async_load_b128(&Alds[buf][ar][ach + 8], asrc + 8);
    bf16x16 v = *reinterpret_cast<const bf16x16*>(
        Bw + (size_t)(k0 + bkr) * GN + n0 + bcc);
#pragma unroll
    for (int i = 0; i < 16; ++i) Bt[buf][bcc + i][bkr] = v[i];
  };

  f32x8 acc[2][4] = {};

  stage(0, 0);
  async_wait0();
  __syncthreads();

  int cur = 0;
  for (int k0 = 0; k0 < GK; k0 += 32) {
    const int nxt = cur ^ 1;
    if (k0 + 32 < GK) {
      stage(nxt, k0 + 32);
      if (k0 + 64 < GK)
        __builtin_prefetch(A + (size_t)(m0 + ar) * GK + k0 + 64, 0, 1);
    }

    bf16x16 af[2], bf_[4];
#pragma unroll
    for (int mt = 0; mt < 2; ++mt)
      af[mt] = load_a_frag(&Alds[cur][mw + mt * 16 + l16][0], halfid);
#pragma unroll
    for (int nt = 0; nt < 4; ++nt)
      bf_[nt] = *reinterpret_cast<const bf16x16*>(
          &Bt[cur][nw + nt * 16 + l16][halfid * 16]);
#pragma unroll
    for (int mt = 0; mt < 2; ++mt)
#pragma unroll
      for (int nt = 0; nt < 4; ++nt)
        acc[mt][nt] = wmma_bf16(af[mt], bf_[nt], acc[mt][nt]);

    async_wait0();
    __syncthreads();
    cur = nxt;
  }

  // epilogue: branch-free, immediate-offset stores (row stride = GN elements)
#pragma unroll
  for (int mt = 0; mt < 2; ++mt)
#pragma unroll
    for (int nt = 0; nt < 4; ++nt) {
      OutT* cp = C + (size_t)(m0 + mw + mt * 16 + halfid * 8) * GN
                   + (n0 + nw + nt * 16 + l16);
#pragma unroll
      for (int i = 0; i < 8; ++i) {
        float v = acc[mt][nt][i];
        if constexpr (sizeof(OutT) == 4) cp[(size_t)i * GN] = v;
        else                             cp[(size_t)i * GN] = f2bf16(v);
      }
    }
}

// ---------------------------------------------------------------------------
// Flash attention per (b,h): q,k,v bf16 in [B,S,H,D] layout.
// 128 threads = 4 waves; each wave owns 16 query rows (64 per WG).
// Key blocks of 32: scores via WMMA, bias add, online softmax, P*V via WMMA.
// ---------------------------------------------------------------------------
__global__ void __launch_bounds__(128)
k_flash_attn(const __bf16* __restrict__ qbf, const __bf16* __restrict__ kbf,
             const __bf16* __restrict__ vbf, const float* __restrict__ relTable,
             __bf16* __restrict__ obf) {
  __shared__ __align__(32) __bf16 Klds[32][DKV];       // [key][d]
  __shared__ __align__(32) __bf16 Vt[DKV][32];         // [d][key]
  __shared__ __align__(32) __bf16 Plds[4][16][32];     // per-wave probs

  const int blk = blockIdx.x;             // ((b*16 + h) * 32 + qtile)
  const int qtile = blk & 31;
  const int h = (blk >> 5) & 15;
  const int b = blk >> 9;
  const int t = threadIdx.x, wave = t >> 5, lane = t & 31;
  const int halfid = lane >> 4, l16 = lane & 15;
  const int qm0 = qtile * 64 + wave * 16;

  // Persistent Q fragments for this wave's 16 rows (K = d 0..31 and 32..63)
  const __bf16* qrow =
      qbf + ((size_t)(b * SEQ + qm0 + l16) * HEADS + h) * DKV;
  const bf16x16 aq0 = load_a_frag(qrow + 0, halfid);
  const bf16x16 aq1 = load_a_frag(qrow + 32, halfid);

  f32x8 O[4] = {};
  float m_i[8], l_i[8];
#pragma unroll
  for (int i = 0; i < 8; ++i) { m_i[i] = -1e30f; l_i[i] = 0.0f; }

  const float* biasRow = relTable + h * RELSPAN + 2047;

  for (int kb = 0; kb < SEQ; kb += 32) {
    __syncthreads();
    { // stage K (row-major, async) and V (transposed, via VGPRs)
      int kr = t >> 2, ch = (t & 3) * 16;
      size_t src = ((size_t)(b * SEQ + kb + kr) * HEADS + h) * DKV + ch;
      const __bf16* ksrc = kbf + src;
      async_load_b128(&Klds[kr][ch], ksrc);
      async_load_b128(&Klds[kr][ch + 8], ksrc + 8);
      bf16x16 vv = *reinterpret_cast<const bf16x16*>(vbf + src);
#pragma unroll
      for (int i = 0; i < 16; ++i) Vt[ch + i][kr] = vv[i];
      if (kb + 32 < SEQ)
        __builtin_prefetch(kbf + src + (size_t)32 * HEADS * DKV, 0, 1);
    }
    async_wait0();
    __syncthreads();

    // scores for two 16-key tiles
    f32x8 s[2];
#pragma unroll
    for (int n = 0; n < 2; ++n) {
      const __bf16* kp = &Klds[n * 16 + l16][halfid * 16];
      bf16x16 bk0 = *reinterpret_cast<const bf16x16*>(kp);        // d 0..31
      bf16x16 bk1 = *reinterpret_cast<const bf16x16*>(kp + 32);   // d 32..63
      f32x8 c = {};
      c = wmma_bf16(aq0, bk0, c);
      c = wmma_bf16(aq1, bk1, c);
#pragma unroll
      for (int i = 0; i < 8; ++i) {
        int qr = qm0 + i + halfid * 8;
        int kc = kb + n * 16 + l16;
        c[i] += biasRow[kc - qr];
      }
      s[n] = c;
    }

    // online softmax (row = i + 8*halfid; reduce across the 16-lane group)
#pragma unroll
    for (int i = 0; i < 8; ++i) {
      float mx = fmaxf(s[0][i], s[1][i]);
#pragma unroll
      for (int off = 8; off >= 1; off >>= 1)
        mx = fmaxf(mx, __shfl_xor(mx, off, 16));
      float mnew  = fmaxf(m_i[i], mx);
      float scale = __expf(m_i[i] - mnew);
      float p0 = __expf(s[0][i] - mnew);
      float p1 = __expf(s[1][i] - mnew);
      float rs = p0 + p1;
#pragma unroll
      for (int off = 8; off >= 1; off >>= 1)
        rs += __shfl_xor(rs, off, 16);
      l_i[i] = l_i[i] * scale + rs;
      m_i[i] = mnew;
      s[0][i] = p0; s[1][i] = p1;
#pragma unroll
      for (int d = 0; d < 4; ++d) O[d][i] *= scale;
    }

    // repack probs C-layout -> A-layout through per-wave LDS
#pragma unroll
    for (int n = 0; n < 2; ++n)
#pragma unroll
      for (int i = 0; i < 8; ++i)
        Plds[wave][i + halfid * 8][n * 16 + l16] = f2bf16(s[n][i]);
    bf16x16 ap = load_a_frag(&Plds[wave][l16][0], halfid);

    // O += P * V  (4 d-tiles of 16)
#pragma unroll
    for (int d = 0; d < 4; ++d) {
      bf16x16 bv =
          *reinterpret_cast<const bf16x16*>(&Vt[d * 16 + l16][halfid * 16]);
      O[d] = wmma_bf16(ap, bv, O[d]);
    }
  }

  // normalize and store bf16 [B,S,H,D] (immediate-offset stores)
  float inv[8];
#pragma unroll
  for (int i = 0; i < 8; ++i) inv[i] = 1.0f / l_i[i];
#pragma unroll
  for (int d = 0; d < 4; ++d) {
    __bf16* op = obf +
        ((size_t)(b * SEQ + qm0 + halfid * 8) * HEADS + h) * DKV + d * 16 + l16;
#pragma unroll
    for (int i = 0; i < 8; ++i)
      op[(size_t)i * HEADS * DKV] = f2bf16(O[d][i] * inv[i]);
  }
}

// ---------------------------------------------------------------------------
// Host launcher
// ---------------------------------------------------------------------------
extern "C" void kernel_launch(void* const* d_in, const int* in_sizes, int n_in,
                              void* d_out, int out_size, void* d_ws,
                              size_t ws_size, hipStream_t stream) {
  const float* X   = (const float*)d_in[0];  // [4,2048,1024]
  const float* Wq  = (const float*)d_in[1];
  const float* Wk  = (const float*)d_in[2];
  const float* Wv  = (const float*)d_in[3];
  const float* Wo  = (const float*)d_in[4];
  const float* tbl = (const float*)d_in[5];  // [32,16]

  float* outAttn = (float*)d_out;                                   // [4,2048,1024]
  float* outBias = outAttn + (size_t)BATCH * SEQ * DMODEL;          // [1,16,S,S]

  const size_t MK = (size_t)BATCH * SEQ * DMODEL;   // 8192*1024
  const size_t KK = (size_t)DMODEL * DMODEL;        // 1024*1024

  char* ws = (char*)d_ws;
  size_t off = 0;
  __bf16* Xbf = (__bf16*)(ws + off); off += MK * 2;
  __bf16* Wqb = (__bf16*)(ws + off); off += KK * 2;
  __bf16* Wkb = (__bf16*)(ws + off); off += KK * 2;
  __bf16* Wvb = (__bf16*)(ws + off); off += KK * 2;
  __bf16* Wob = (__bf16*)(ws + off); off += KK * 2;
  __bf16* Qb  = (__bf16*)(ws + off); off += MK * 2;
  __bf16* Kb  = (__bf16*)(ws + off); off += MK * 2;
  __bf16* Vb  = (__bf16*)(ws + off); off += MK * 2;
  __bf16* Ob  = (__bf16*)(ws + off); off += MK * 2;
  float*  relT = (float*)(ws + off); off += (size_t)HEADS * RELSPAN * 4;

  // 1) bf16 conversions
  k_f32_to_bf16<<<2048, 256, 0, stream>>>(X,  Xbf, (int)MK);
  k_f32_to_bf16<<<1024, 256, 0, stream>>>(Wq, Wqb, (int)KK);
  k_f32_to_bf16<<<1024, 256, 0, stream>>>(Wk, Wkb, (int)KK);
  k_f32_to_bf16<<<1024, 256, 0, stream>>>(Wv, Wvb, (int)KK);
  k_f32_to_bf16<<<1024, 256, 0, stream>>>(Wo, Wob, (int)KK);

  // 2) relative-position bias table + full pos_bias output
  k_build_rel_table<<<(HEADS * RELSPAN + 255) / 256, 256, 0, stream>>>(tbl, relT);
  k_write_pos_bias<<<(HEADS * SEQ * SEQ / 4) / 256, 256, 0, stream>>>(relT, outBias);

  // 3) Q/K/V projections (bf16 out)
  dim3 gg(GM / 128, GN / 128);
  k_gemm_tpl<__bf16><<<gg, 256, 0, stream>>>(Xbf, Wqb, Qb);
  k_gemm_tpl<__bf16><<<gg, 256, 0, stream>>>(Xbf, Wkb, Kb);
  k_gemm_tpl<__bf16><<<gg, 256, 0, stream>>>(Xbf, Wvb, Vb);

  // 4) flash attention: B*H*(S/64) workgroups of 128 threads
  k_flash_attn<<<BATCH * HEADS * (SEQ / 64), 128, 0, stream>>>(Qb, Kb, Vb, relT, Ob);

  // 5) output projection (f32 out into d_out)
  k_gemm_tpl<float><<<gg, 256, 0, stream>>>(Ob, Wob, outAttn);
}
